// VAE_42477226557479
// MI455X (gfx1250) — matrix operand order repaired and checked by reading, not compiled, not run on hardware
//
#include <hip/hip_runtime.h>
#include <hip/hip_bf16.h>
#include <math.h>

// Problem constants (match reference)
namespace {
constexpr int kB   = 512;
constexpr int kT   = 512;
constexpr int kE   = 100;
constexpr int kH   = 100;
constexpr int kNZ  = 50;
constexpr int kNIN = 2000;
constexpr int kG   = 4 * kH;       // 400 gate columns
constexpr int kKP  = 128;          // K padded to 4 WMMA chunks of 32
constexpr int kNT  = kG / 16;      // 25 n-tiles
constexpr int kCH  = kKP / 32;     // 4 k-chunks
}

typedef __attribute__((ext_vector_type(16))) _Float16 v16h;
typedef __attribute__((ext_vector_type(8)))  _Float16 v8h;
typedef __attribute__((ext_vector_type(8)))  float    v8f;

__device__ __forceinline__ float sigmoidf_(float x) { return 1.f / (1.f + __expf(-x)); }

// ---------------------------------------------------------------------------
// WMMA GEMM, f16 operands / f32 accumulate (V_WMMA_F32_16X16X32_F16).
//
// A is stored row-major with row stride kKP (=128) halves, pad region zeroed,
// so a lane's A fragment (ISA 7.12.2: two contiguous 8-half runs at
// k = kb..kb+7 and kb+16..kb+23, kb = lane<16 ? 0 : 8) is two aligned
// global_load_b128 with NO bounds checks.
//
// B is pre-packed into fragment order: Bp[((chunk*kNT + ntile)*32 + lane)*16]
// holds that lane's 16 halves contiguously (32 B, 32 B-aligned) -> two
// global_load_b128 per chunk.
//
// K loop is fully unrolled (4 chunks); zero padding makes the extra K range
// a no-op numerically. One wave per 16x16 tile; no EXEC divergence anywhere.
// ---------------------------------------------------------------------------
__global__ void wmma_gemm(const _Float16* __restrict__ A,
                          const _Float16* __restrict__ Bp,
                          const float* __restrict__ Cinit, long long cRowStride,
                          const float* __restrict__ bias,
                          float* __restrict__ Out, long long outRowStride) {
  int lane  = threadIdx.x & 31;
  int mBase = blockIdx.x * 16;
  int nt    = blockIdx.y;
  int n     = nt * 16 + (lane & 15);
  int mOff  = (lane < 16) ? 0 : 8;

  const _Float16* arow = A + (size_t)(mBase + (lane & 15)) * kKP + ((lane < 16) ? 0 : 8);
  const v16h*     bfrag = (const v16h*)Bp + (size_t)nt * 32 + lane;

  __builtin_prefetch(bfrag, 0, 3);   // global_prefetch_b8 on the weight stream

  v8f c;
  if (Cinit) {
#pragma unroll
    for (int r = 0; r < 8; ++r)
      c[r] = Cinit[(long long)(mBase + r + mOff) * cRowStride + n];
  } else {
    float bv = bias ? bias[n] : 0.f;
#pragma unroll
    for (int r = 0; r < 8; ++r) c[r] = bv;
  }

#pragma unroll
  for (int chunk = 0; chunk < kCH; ++chunk) {
    v8h lo = *(const v8h*)(arow + chunk * 32);        // k = kBase+kb .. +7
    v8h hi = *(const v8h*)(arow + chunk * 32 + 16);   // k = kBase+kb+16 .. +23
    v16h a = __builtin_shufflevector(lo, hi, 0, 1, 2, 3, 4, 5, 6, 7,
                                     8, 9, 10, 11, 12, 13, 14, 15);
    v16h b = bfrag[(size_t)chunk * (kNT * 32)];
    c = __builtin_amdgcn_wmma_f32_16x16x32_f16(false, a, false, b, (short)0, c,
                                               false, false);
  }

#pragma unroll
  for (int r = 0; r < 8; ++r)
    Out[(long long)(mBase + r + mOff) * outRowStride + n] = c[r];
}

// ---------------------------------------------------------------------------
// Pack a K x 400 slice of W (f32, row-major, rows kRow0..kRow0+Kmax) into the
// per-lane WMMA B-fragment layout described above. 51200 halves per matrix.
// ---------------------------------------------------------------------------
__global__ void pack_b(const float* __restrict__ W, int kRow0, int Kmax,
                       _Float16* __restrict__ dst) {
  int idx = blockIdx.x * blockDim.x + threadIdx.x;
  if (idx >= kCH * kNT * 32 * 16) return;
  int i     = idx & 15;
  int lane  = (idx >> 4) & 31;
  int nt    = (idx >> 9) % kNT;
  int chunk = idx / (16 * 32 * kNT);
  int n  = nt * 16 + (lane & 15);
  int kb = (lane < 16) ? 0 : 8;
  int k  = chunk * 32 + kb + (i & 7) + ((i & 8) ? 16 : 0);
  float v = (k < Kmax) ? W[(size_t)(kRow0 + k) * kG + n] : 0.f;
  dst[idx] = (_Float16)v;
}

__global__ void zero_f32(float* p, int n) {
  int idx = blockIdx.x * blockDim.x + threadIdx.x;
  if (idx < n) p[idx] = 0.f;
}
__global__ void zero_f16(_Float16* p, int n) {
  int idx = blockIdx.x * blockDim.x + threadIdx.x;
  if (idx < n) p[idx] = (_Float16)0.f;
}

// Embedding gather (f32 table -> f16, rows padded to kKP with zeros). Also
// writes the length-reversed sequence for the backward direction.
__global__ void embed_gather(const int* __restrict__ Xin, const int* __restrict__ len,
                             const float* __restrict__ emb,
                             _Float16* __restrict__ xH, _Float16* __restrict__ xrH) {
  int idx = blockIdx.x * blockDim.x + threadIdx.x;
  if (idx >= kB * kT) return;
  int b = idx / kT, t = idx % kT;
  int L  = len[b];
  int rf = Xin[idx];
  int tr = (t < L) ? (L - 1 - t) : t;
  int rb = Xin[b * kT + tr];
  const float* ef = emb + (size_t)rf * kE;
  const float* eb = emb + (size_t)rb * kE;
  _Float16* of = xH  + (size_t)idx * kKP;
  _Float16* ob = xrH + (size_t)idx * kKP;
  for (int e = 0; e < kE; ++e) {
    of[e] = (_Float16)ef[e];
    ob[e] = (_Float16)eb[e];
  }
  for (int e = kE; e < kKP; ++e) { of[e] = (_Float16)0.f; ob[e] = (_Float16)0.f; }
}

// Gate nonlinearities + state update + masked running-mean accumulation.
// Masked time-mean is permutation invariant, so the backward direction needs
// no output un-reversal: just accumulate the masked h_new.
__global__ void lstm_point(const float* __restrict__ G, const int* __restrict__ len,
                           int t, float* __restrict__ h, float* __restrict__ c,
                           _Float16* __restrict__ hH, float* __restrict__ sum) {
  int idx = blockIdx.x * blockDim.x + threadIdx.x;
  if (idx >= kB * kH) return;
  int b = idx / kH, j = idx % kH;
  const float* g = G + (size_t)b * kG;
  float gi = g[j], gj = g[kH + j], gf = g[2 * kH + j], go = g[3 * kH + j];
  float cn = c[idx] * sigmoidf_(gf + 1.f) + sigmoidf_(gi) * tanhf(gj);
  float hn = tanhf(cn) * sigmoidf_(go);
  if (t < len[b]) {
    c[idx] = cn;
    h[idx] = hn;
    sum[idx] += hn;               // out = where(mask, h_new, 0); summed over t
  }
  hH[(size_t)b * kKP + j] = (_Float16)h[idx];  // padded f16 copy -> next WMMA A
}

// pooled = concat([mean_bw, mean_fw]) = sums * (1/len)
__global__ void pool_kernel(const float* __restrict__ sumFw, const float* __restrict__ sumBw,
                            const int* __restrict__ len, float* __restrict__ pooled) {
  int idx = blockIdx.x * blockDim.x + threadIdx.x;
  if (idx >= kB * 2 * kH) return;
  int b = idx / (2 * kH), j = idx % (2 * kH);
  float il = 1.f / (float)len[b];
  float v  = (j < kH) ? sumBw[b * kH + j] : sumFw[b * kH + (j - kH)];
  pooled[idx] = v * il;
}

// Small dense projection: Out[m,n] = A[m,:K] @ W[:K,n] (+ bias)
__global__ void proj_kernel(const float* __restrict__ A, const float* __restrict__ W,
                            const float* __restrict__ bias, float* __restrict__ Out,
                            int M, int K, int N) {
  int idx = blockIdx.x * blockDim.x + threadIdx.x;
  if (idx >= M * N) return;
  int m = idx / N, n = idx % N;
  float acc = bias ? bias[n] : 0.f;
  const float* a = A + (size_t)m * K;
  for (int k = 0; k < K; ++k) acc = fmaf(a[k], W[(size_t)k * N + n], acc);
  Out[idx] = acc;
}

// Batchnorm over axis 0 (population variance, eps = 1e-3)
__global__ void bn_stats(const float* __restrict__ X, int M, int N,
                         float* __restrict__ mean, float* __restrict__ rstd) {
  int n = blockIdx.x * blockDim.x + threadIdx.x;
  if (n >= N) return;
  float s = 0.f;
  for (int m = 0; m < M; ++m) s += X[(size_t)m * N + n];
  float mu = s / (float)M;
  float v = 0.f;
  for (int m = 0; m < M; ++m) { float d = X[(size_t)m * N + n] - mu; v = fmaf(d, d, v); }
  mean[n] = mu;
  rstd[n] = rsqrtf(v / (float)M + 1e-3f);
}

__global__ void bn_apply(float* __restrict__ X, int M, int N,
                         const float* __restrict__ mean, const float* __restrict__ rstd) {
  int idx = blockIdx.x * blockDim.x + threadIdx.x;
  if (idx >= M * N) return;
  int n = idx % N;
  X[idx] = (X[idx] - mean[n]) * rstd[n];
}

// z = bn(z_mean) + sqrt(exp(bn(z_lsig))) * eps   (sqrt(exp(x)) = exp(0.5x))
__global__ void z_combine(const float* __restrict__ zm, const float* __restrict__ zl,
                          const float* __restrict__ eps, float* __restrict__ z) {
  int idx = blockIdx.x * blockDim.x + threadIdx.x;
  if (idx >= kB * kNZ) return;
  int n = idx % kNZ;
  z[idx] = zm[idx] + __expf(0.5f * zl[idx]) * eps[n];
}

__global__ void softmax_rows(float* __restrict__ X, int M, int N) {
  int m = blockIdx.x * blockDim.x + threadIdx.x;
  if (m >= M) return;
  float* row = X + (size_t)m * N;
  float mx = -3.4e38f;
  for (int n = 0; n < N; ++n) mx = fmaxf(mx, row[n]);
  float s = 0.f;
  for (int n = 0; n < N; ++n) { float e = __expf(row[n] - mx); row[n] = e; s += e; }
  float inv = 1.f / s;
  for (int n = 0; n < N; ++n) row[n] *= inv;
}

// ---------------------------------------------------------------------------
extern "C" void kernel_launch(void* const* d_in, const int* in_sizes, int n_in,
                              void* d_out, int out_size, void* d_ws, size_t ws_size,
                              hipStream_t stream) {
  const int*   Xin   = (const int*)d_in[0];
  const int*   Lens  = (const int*)d_in[1];
  const float* Emb   = (const float*)d_in[2];
  const float* Wfw   = (const float*)d_in[3];
  const float* bfw   = (const float*)d_in[4];
  const float* Wbw   = (const float*)d_in[5];
  const float* bbw   = (const float*)d_in[6];
  const float* Wmean = (const float*)d_in[7];
  const float* bmean = (const float*)d_in[8];
  const float* Wlsig = (const float*)d_in[9];
  const float* blsig = (const float*)d_in[10];
  const float* Wgen  = (const float*)d_in[11];
  const float* EpsV  = (const float*)d_in[12];
  float* Out = (float*)d_out;

  // Workspace carve-out (~1.0 GB; trivial vs 432 GiB HBM).
  char* ws = (char*)d_ws;
  size_t off = 0;
  auto alloc = [&](size_t bytes) -> char* {
    char* p = ws + off;
    off = (off + bytes + 255) & ~(size_t)255;
    return p;
  };
  const int packElems = kCH * kNT * 32 * 16;  // 51200 halves per packed matrix
  _Float16* BpXfw = (_Float16*)alloc((size_t)packElems * sizeof(_Float16));
  _Float16* BpHfw = (_Float16*)alloc((size_t)packElems * sizeof(_Float16));
  _Float16* BpXbw = (_Float16*)alloc((size_t)packElems * sizeof(_Float16));
  _Float16* BpHbw = (_Float16*)alloc((size_t)packElems * sizeof(_Float16));
  _Float16* xH    = (_Float16*)alloc((size_t)kB * kT * kKP * sizeof(_Float16));
  _Float16* xrH   = (_Float16*)alloc((size_t)kB * kT * kKP * sizeof(_Float16));
  float*    XgFw  = (float*)alloc((size_t)kB * kT * kG * sizeof(float));
  float*    XgBw  = (float*)alloc((size_t)kB * kT * kG * sizeof(float));
  float*    hFw   = (float*)alloc((size_t)kB * kH * sizeof(float));
  float*    cFw   = (float*)alloc((size_t)kB * kH * sizeof(float));
  float*    sumFw = (float*)alloc((size_t)kB * kH * sizeof(float));
  _Float16* hFwH  = (_Float16*)alloc((size_t)kB * kKP * sizeof(_Float16));
  float*    hBw   = (float*)alloc((size_t)kB * kH * sizeof(float));
  float*    cBw   = (float*)alloc((size_t)kB * kH * sizeof(float));
  float*    sumBw = (float*)alloc((size_t)kB * kH * sizeof(float));
  _Float16* hBwH  = (_Float16*)alloc((size_t)kB * kKP * sizeof(_Float16));
  float*    GFw   = (float*)alloc((size_t)kB * kG * sizeof(float));
  float*    GBw   = (float*)alloc((size_t)kB * kG * sizeof(float));
  float*    pooled= (float*)alloc((size_t)kB * 2 * kH * sizeof(float));
  float*    zm    = (float*)alloc((size_t)kB * kNZ * sizeof(float));
  float*    zl    = (float*)alloc((size_t)kB * kNZ * sizeof(float));
  float*    zbuf  = (float*)alloc((size_t)kB * kNZ * sizeof(float));
  float*    cMean = (float*)alloc((size_t)kNIN * sizeof(float));
  float*    cRstd = (float*)alloc((size_t)kNIN * sizeof(float));
  (void)ws_size; (void)in_sizes; (void)n_in; (void)out_size;

  const int BT = kB * kT;

  // 1. Pack weights into WMMA B-fragment layout (input rows 0..99, recurrent
  //    rows 100..199, K zero-padded to 128).
  pack_b<<<(packElems + 255) / 256, 256, 0, stream>>>(Wfw, 0,  kE, BpXfw);
  pack_b<<<(packElems + 255) / 256, 256, 0, stream>>>(Wfw, kE, kH, BpHfw);
  pack_b<<<(packElems + 255) / 256, 256, 0, stream>>>(Wbw, 0,  kE, BpXbw);
  pack_b<<<(packElems + 255) / 256, 256, 0, stream>>>(Wbw, kE, kH, BpHbw);

  // 2. Zero LSTM state + masked-mean accumulators (+ padded f16 h copies)
  zero_f32<<<(kB * kH + 255) / 256, 256, 0, stream>>>(hFw, kB * kH);
  zero_f32<<<(kB * kH + 255) / 256, 256, 0, stream>>>(cFw, kB * kH);
  zero_f32<<<(kB * kH + 255) / 256, 256, 0, stream>>>(sumFw, kB * kH);
  zero_f16<<<(kB * kKP + 255) / 256, 256, 0, stream>>>(hFwH, kB * kKP);
  zero_f32<<<(kB * kH + 255) / 256, 256, 0, stream>>>(hBw, kB * kH);
  zero_f32<<<(kB * kH + 255) / 256, 256, 0, stream>>>(cBw, kB * kH);
  zero_f32<<<(kB * kH + 255) / 256, 256, 0, stream>>>(sumBw, kB * kH);
  zero_f16<<<(kB * kKP + 255) / 256, 256, 0, stream>>>(hBwH, kB * kKP);

  // 3. Embedding gather (forward + length-reversed sequence)
  embed_gather<<<(BT + 255) / 256, 256, 0, stream>>>(Xin, Lens, Emb, xH, xrH);

  // 4. Hoisted input projection (WMMA, M=262144): Xg = x @ W[0:E,:] + b
  wmma_gemm<<<dim3(BT / 16, kNT), 32, 0, stream>>>(
      xH, BpXfw, nullptr, 0, bfw, XgFw, kG);
  wmma_gemm<<<dim3(BT / 16, kNT), 32, 0, stream>>>(
      xrH, BpXbw, nullptr, 0, bbw, XgBw, kG);

  // 5. Serial scan: only the recurrent GEMM (K=H) + gate pointwise remain.
  //    Xg[:, t, :] (row stride T*kG) seeds the WMMA accumulator.
  for (int t = 0; t < kT; ++t) {
    wmma_gemm<<<dim3(kB / 16, kNT), 32, 0, stream>>>(
        hFwH, BpHfw, XgFw + (size_t)t * kG, (long long)kT * kG, nullptr, GFw, kG);
    wmma_gemm<<<dim3(kB / 16, kNT), 32, 0, stream>>>(
        hBwH, BpHbw, XgBw + (size_t)t * kG, (long long)kT * kG, nullptr, GBw, kG);
    lstm_point<<<(kB * kH + 255) / 256, 256, 0, stream>>>(GFw, Lens, t, hFw, cFw, hFwH, sumFw);
    lstm_point<<<(kB * kH + 255) / 256, 256, 0, stream>>>(GBw, Lens, t, hBw, cBw, hBwH, sumBw);
  }

  // 6. Masked mean pool -> pooled = [mean_bw, mean_fw]
  pool_kernel<<<(kB * 2 * kH + 255) / 256, 256, 0, stream>>>(sumFw, sumBw, Lens, pooled);

  // 7. z_mean / z_log_sigma_sq projections + batchnorm
  proj_kernel<<<(kB * kNZ + 255) / 256, 256, 0, stream>>>(pooled, Wmean, bmean, zm, kB, 2 * kH, kNZ);
  proj_kernel<<<(kB * kNZ + 255) / 256, 256, 0, stream>>>(pooled, Wlsig, blsig, zl, kB, 2 * kH, kNZ);
  bn_stats<<<1, 64, 0, stream>>>(zm, kB, kNZ, cMean, cRstd);
  bn_apply<<<(kB * kNZ + 255) / 256, 256, 0, stream>>>(zm, kB, kNZ, cMean, cRstd);
  bn_stats<<<1, 64, 0, stream>>>(zl, kB, kNZ, cMean, cRstd);
  bn_apply<<<(kB * kNZ + 255) / 256, 256, 0, stream>>>(zl, kB, kNZ, cMean, cRstd);

  // 8. Reparameterize + softmax -> theta (in zbuf)
  z_combine<<<(kB * kNZ + 255) / 256, 256, 0, stream>>>(zm, zl, EpsV, zbuf);
  softmax_rows<<<(kB + 255) / 256, 256, 0, stream>>>(zbuf, kB, kNZ);

  // 9. Decoder: r = theta @ W_gen -> bn -> softmax, in d_out
  proj_kernel<<<(kB * kNIN + 255) / 256, 256, 0, stream>>>(zbuf, Wgen, nullptr, Out, kB, kNZ, kNIN);
  bn_stats<<<(kNIN + 255) / 256, 256, 0, stream>>>(Out, kB, kNIN, cMean, cRstd);
  bn_apply<<<(kB * kNIN + 255) / 256, 256, 0, stream>>>(Out, kB, kNIN, cMean, cRstd);
  softmax_rows<<<(kB + 255) / 256, 256, 0, stream>>>(Out, kB, kNIN);
}
